// HANHeterogeneous_89584427860364
// MI455X (gfx1250) — matrix operand order repaired and checked by reading, not compiled, not run on hardware
//
#include <hip/hip_runtime.h>
#include <hip/hip_bf16.h>

#define N_NODES   100000
#define N_EDGES   1000000
#define N_REL     3
#define N_LAYERS  2
#define HDIM      64
#define N_CLASSES 8
#define N_GRAPHS  64
#define NEG_SLOPE 0.2f

static_assert(N_NODES % 32 == 0, "row-tile coverage must be exact (no store guards)");
static_assert((N_NODES * HDIM) % 4 == 0, "float4 elementwise kernels");

typedef __attribute__((ext_vector_type(16))) __bf16 v16bf;
typedef __attribute__((ext_vector_type(8)))  float  v8f;

__device__ __forceinline__ float lrelu(float v) { return v > 0.f ? v : NEG_SLOPE * v; }

// order-preserving float <-> uint key for atomicMax-based segment max
__device__ __forceinline__ unsigned fkey(float f) {
    unsigned u = __float_as_uint(f);
    return (u & 0x80000000u) ? ~u : (u | 0x80000000u);
}
__device__ __forceinline__ float keyf(unsigned k) {
    return __uint_as_float((k & 0x80000000u) ? (k & 0x7FFFFFFFu) : ~k);
}

// ---------------------------------------------------------------------------
// Pack W[64,64] (row-major f32) -> Wt[col*64+k] (col-major bf16), 8 KB.
// Makes each lane's B fragment a single contiguous 32-byte (v16bf) load.
// ---------------------------------------------------------------------------
__global__ void __launch_bounds__(256)
pack_w_kernel(const float* __restrict__ W, __bf16* __restrict__ Wt)
{
    for (int i = threadIdx.x; i < HDIM * HDIM; i += 256) {
        int k = i >> 6, c = i & 63;
        Wt[c * HDIM + k] = (__bf16)W[k * HDIM + c];
    }
}

// ---------------------------------------------------------------------------
// GEMM: C[N,64] = A[N,64] @ W[64,64] + bias. bf16 operands, f32 accumulate.
// One wave owns a 16x16 tile; block = 8 waves = 32 rows x 64 cols. K=64 in
// two v_wmma_f32_16x16x32_bf16 steps. Grid covers N exactly -> no guards.
// ---------------------------------------------------------------------------
__global__ void __launch_bounds__(256)
proj_wmma(const float* __restrict__ A, const __bf16* __restrict__ Wt,
          const float* __restrict__ bias, float* __restrict__ C)
{
    const int lane = threadIdx.x & 31;
    const int wave = threadIdx.x >> 5;
    const int row0 = (blockIdx.x * 2 + (wave >> 2)) * 16;
    const int colTile = (wave & 3) * 16;
    const int l15 = lane & 15;
    const int hi  = lane >> 4;
    const int col = colTile + l15;

    float bb = bias[col];
    v8f acc;
#pragma unroll
    for (int v = 0; v < 8; ++v) acc[v] = bb;

    const float*  Arow = A  + (long)(row0 + l15) * HDIM;
    const __bf16* Wcol = Wt + col * HDIM;

#pragma unroll
    for (int k0 = 0; k0 < HDIM; k0 += 32) {
        // A fragment: lanes<16 hold K=0..15, lanes>=16 K=16..31
        // -> two contiguous 8-float chunks per lane (vectorizes to b128 loads)
        float abuf[16];
#pragma unroll
        for (int j = 0; j < 8; ++j) abuf[j]     = Arow[k0 + hi * 16 + j];
#pragma unroll
        for (int j = 0; j < 8; ++j) abuf[8 + j] = Arow[k0 + hi * 16 + 8 + j];
        v16bf af;
#pragma unroll
        for (int j = 0; j < 16; ++j) af[j] = (__bf16)abuf[j];

        // B fragment: lanes<16 K=0..15, lanes>=16 K=16..31 of column `col`,
        // contiguous packed bf16 -> one aligned 32-byte vector load
        v16bf wf = *(const v16bf*)(Wcol + k0 + hi * 16);

        acc = __builtin_amdgcn_wmma_f32_16x16x32_bf16(false, af, false, wf,
                                                      (short)0, acc, false, false);
    }
#pragma unroll
    for (int v = 0; v < 8; ++v)
        C[(long)(row0 + v + hi * 8) * HDIM + col] = acc[v];
}

// ---------------------------------------------------------------------------
// Semantic attention GEMM: colsum[c] += sum_rows tanh((A @ Wk + bk))[.,c]
// ---------------------------------------------------------------------------
__global__ void __launch_bounds__(256)
semantic_wmma(const float* __restrict__ A, const __bf16* __restrict__ Wkt,
              const float* __restrict__ bk, float* __restrict__ colsum)
{
    const int lane = threadIdx.x & 31;
    const int wave = threadIdx.x >> 5;
    const int row0 = (blockIdx.x * 2 + (wave >> 2)) * 16;
    const int colTile = (wave & 3) * 16;
    const int l15 = lane & 15;
    const int hi  = lane >> 4;
    const int col = colTile + l15;

    float bb = bk[col];
    v8f acc;
#pragma unroll
    for (int v = 0; v < 8; ++v) acc[v] = bb;

    const float*  Arow = A   + (long)(row0 + l15) * HDIM;
    const __bf16* Wcol = Wkt + col * HDIM;

#pragma unroll
    for (int k0 = 0; k0 < HDIM; k0 += 32) {
        float abuf[16];
#pragma unroll
        for (int j = 0; j < 8; ++j) abuf[j]     = Arow[k0 + hi * 16 + j];
#pragma unroll
        for (int j = 0; j < 8; ++j) abuf[8 + j] = Arow[k0 + hi * 16 + 8 + j];
        v16bf af;
#pragma unroll
        for (int j = 0; j < 16; ++j) af[j] = (__bf16)abuf[j];

        v16bf wf = *(const v16bf*)(Wcol + k0 + hi * 16);

        acc = __builtin_amdgcn_wmma_f32_16x16x32_bf16(false, af, false, wf,
                                                      (short)0, acc, false, false);
    }
    float partial = 0.f;
#pragma unroll
    for (int v = 0; v < 8; ++v) partial += tanhf(acc[v]);
    partial += __shfl_down(partial, 16);   // join the two M-halves of the column
    if (hi == 0) atomicAdd(&colsum[col], partial);
}

// ---------------------------------------------------------------------------
// Per-node attention logits: alpha_src/dst[r][n] = h[n] . a_{s,d}[r]
// one wave per node, float2 per lane, shfl reduction
// ---------------------------------------------------------------------------
__global__ void __launch_bounds__(256)
alphas_kernel(const float* __restrict__ h, const float* __restrict__ a_s,
              const float* __restrict__ a_d, float* __restrict__ asrc,
              float* __restrict__ adst, int n)
{
    int wave = threadIdx.x >> 5, lane = threadIdx.x & 31;
    int node = blockIdx.x * 8 + wave;
    if (node >= n) return;
    float2 hv = ((const float2*)(h + (long)node * HDIM))[lane];
#pragma unroll
    for (int r = 0; r < N_REL; ++r) {
        float2 as = ((const float2*)(a_s + r * HDIM))[lane];
        float2 ad = ((const float2*)(a_d + r * HDIM))[lane];
        float ps = hv.x * as.x + hv.y * as.y;
        float pd = hv.x * ad.x + hv.y * ad.y;
#pragma unroll
        for (int off = 16; off; off >>= 1) {
            ps += __shfl_down(ps, off);
            pd += __shfl_down(pd, off);
        }
        if (lane == 0) {
            asrc[(long)r * n + node] = ps;
            adst[(long)r * n + node] = pd;
        }
    }
}

__global__ void emax_kernel(const int* __restrict__ ei, const float* __restrict__ asrc,
                            const float* __restrict__ adst, unsigned* __restrict__ mkey,
                            int n, int e)
{
    int i = blockIdx.x * blockDim.x + threadIdx.x;
    int r = blockIdx.y;
    if (i >= e) return;
    int src = ei[((long)r * 2 + 0) * e + i];
    int dst = ei[((long)r * 2 + 1) * e + i];
    float ev = lrelu(asrc[(long)r * n + src] + adst[(long)r * n + dst]);
    atomicMax(&mkey[(long)r * n + dst], fkey(ev));
}

__global__ void esum_kernel(const int* __restrict__ ei, const float* __restrict__ asrc,
                            const float* __restrict__ adst, const unsigned* __restrict__ mkey,
                            float* __restrict__ ssum, float* __restrict__ pbuf,
                            int n, int e)
{
    int i = blockIdx.x * blockDim.x + threadIdx.x;
    int r = blockIdx.y;
    if (i >= e) return;
    int src = ei[((long)r * 2 + 0) * e + i];
    int dst = ei[((long)r * 2 + 1) * e + i];
    float ev = lrelu(asrc[(long)r * n + src] + adst[(long)r * n + dst]);
    float p  = __expf(ev - keyf(mkey[(long)r * n + dst]));
    pbuf[(long)r * e + i] = p;
    atomicAdd(&ssum[(long)r * n + dst], p);
}

// weighted aggregation: wave per edge, float2 gather per lane, f32 atomics
__global__ void __launch_bounds__(256)
agg_kernel(const int* __restrict__ ei, const float* __restrict__ h,
           const float* __restrict__ pbuf, const float* __restrict__ ssum,
           float* __restrict__ outs, int n, int e)
{
    int wave = threadIdx.x >> 5, lane = threadIdx.x & 31;
    int i = blockIdx.x * 8 + wave;
    int r = blockIdx.y;
    if (i >= e) return;
    int src = ei[((long)r * 2 + 0) * e + i];
    int dst = ei[((long)r * 2 + 1) * e + i];
    const float* hs = h + (long)src * HDIM;
    __builtin_prefetch(hs, 0, 0);          // global_prefetch_b8: warm gather row
    float w = pbuf[(long)r * e + i] / (ssum[(long)r * n + dst] + 1e-16f);
    float* od = outs + ((long)r * n + dst) * HDIM;
    float2 hv = ((const float2*)hs)[lane];
    atomicAdd(&od[2 * lane],     hv.x * w);
    atomicAdd(&od[2 * lane + 1], hv.y * w);
}

__global__ void relu4_kernel(float4* __restrict__ x, long n4)
{
    long i = (long)blockIdx.x * blockDim.x + threadIdx.x;
    if (i >= n4) return;
    float4 v = x[i];
    v.x = fmaxf(v.x, 0.f); v.y = fmaxf(v.y, 0.f);
    v.z = fmaxf(v.z, 0.f); v.w = fmaxf(v.w, 0.f);
    x[i] = v;
}

__global__ void beta_kernel(const float* __restrict__ colsum, const float* __restrict__ q,
                            float* __restrict__ beta, int n)
{
    __shared__ float red[HDIM];
    __shared__ float sc[N_REL];
    int t = threadIdx.x;   // 64 threads
    for (int r = 0; r < N_REL; ++r) {
        red[t] = q[t] * (colsum[r * HDIM + t] / (float)n);
        __syncthreads();
        if (t == 0) {
            float s = 0.f;
            for (int j = 0; j < HDIM; ++j) s += red[j];
            sc[r] = s;
        }
        __syncthreads();
    }
    if (t == 0) {
        float m = fmaxf(sc[0], fmaxf(sc[1], sc[2]));
        float e0 = __expf(sc[0] - m), e1 = __expf(sc[1] - m), e2 = __expf(sc[2] - m);
        float s = e0 + e1 + e2;
        beta[0] = e0 / s; beta[1] = e1 / s; beta[2] = e2 / s;
    }
}

__global__ void combine_kernel(const float4* __restrict__ outs, const float* __restrict__ beta,
                               float4* __restrict__ h, long tot4, int dorelu)
{
    long i = (long)blockIdx.x * blockDim.x + threadIdx.x;
    if (i >= tot4) return;
    float b0 = beta[0], b1 = beta[1], b2 = beta[2];
    float4 a = outs[i], b = outs[tot4 + i], c = outs[2 * tot4 + i];
    float4 v;
    v.x = b0 * a.x + b1 * b.x + b2 * c.x;
    v.y = b0 * a.y + b1 * b.y + b2 * c.y;
    v.z = b0 * a.z + b1 * b.z + b2 * c.z;
    v.w = b0 * a.w + b1 * b.w + b2 * c.w;
    if (dorelu) {
        v.x = fmaxf(v.x, 0.f); v.y = fmaxf(v.y, 0.f);
        v.z = fmaxf(v.z, 0.f); v.w = fmaxf(v.w, 0.f);
    }
    h[i] = v;
}

__global__ void __launch_bounds__(256)
pool_kernel(const float* __restrict__ h, const int* __restrict__ batch,
            float* __restrict__ pooled, float* __restrict__ counts, int n)
{
    int wave = threadIdx.x >> 5, lane = threadIdx.x & 31;
    int node = blockIdx.x * 8 + wave;
    if (node >= n) return;
    int g = batch[node];
    float2 hv = ((const float2*)(h + (long)node * HDIM))[lane];
    atomicAdd(&pooled[g * HDIM + 2 * lane],     hv.x);
    atomicAdd(&pooled[g * HDIM + 2 * lane + 1], hv.y);
    if (lane == 0) atomicAdd(&counts[g], 1.0f);
}

__global__ void classify_kernel(const float* __restrict__ pooled, const float* __restrict__ counts,
                                const float* __restrict__ Wout, const float* __restrict__ bout,
                                float* __restrict__ out)
{
    int t = threadIdx.x;   // 512 threads
    if (t >= N_GRAPHS * N_CLASSES) return;
    int g = t / N_CLASSES, c = t % N_CLASSES;
    float cnt = fmaxf(counts[g], 1.f);
    float s = bout[c];
    for (int k = 0; k < HDIM; ++k)
        s += (pooled[g * HDIM + k] / cnt) * Wout[k * N_CLASSES + c];
    out[t] = s;
}

static inline long align64f(long n) { return (n + 63) & ~63L; }   // 256B-aligned carves

extern "C" void kernel_launch(void* const* d_in, const int* in_sizes, int n_in,
                              void* d_out, int out_size, void* d_ws, size_t ws_size,
                              hipStream_t stream)
{
    (void)in_sizes; (void)n_in; (void)out_size; (void)ws_size;
    const float* x      = (const float*)d_in[0];
    const int*   ei     = (const int*)d_in[1];
    const int*   batch  = (const int*)d_in[2];
    const float* W_proj = (const float*)d_in[3];
    const float* b_proj = (const float*)d_in[4];
    const float* a_src  = (const float*)d_in[5];
    const float* a_dst  = (const float*)d_in[6];
    const float* Wk     = (const float*)d_in[7];
    const float* bk     = (const float*)d_in[8];
    const float* q      = (const float*)d_in[9];
    const float* W_out  = (const float*)d_in[10];
    const float* b_out  = (const float*)d_in[11];

    const int N = N_NODES, E = N_EDGES;

    float* ws      = (float*)d_ws;
    float* hproj   = ws;              ws += align64f((long)N * HDIM);
    float* hbuf    = ws;              ws += align64f((long)N * HDIM);
    float* outs    = ws;              ws += align64f((long)N_REL * N * HDIM);
    float* asrc    = ws;              ws += align64f((long)N_REL * N);
    float* adst    = ws;              ws += align64f((long)N_REL * N);
    unsigned* mkey = (unsigned*)ws;   ws += align64f((long)N_REL * N);
    float* ssum    = ws;              ws += align64f((long)N_REL * N);
    float* pbuf    = ws;              ws += align64f((long)N_REL * E);
    float* colsum  = ws;              ws += align64f(N_REL * HDIM);
    float* beta    = ws;              ws += 64;
    float* pooled  = ws;              ws += align64f(N_GRAPHS * HDIM);
    float* counts  = ws;              ws += 64;
    __bf16* wt     = (__bf16*)ws;     ws += align64f(HDIM * HDIM / 2);   // 8 KB packed
    __bf16* wkt    = (__bf16*)ws;     ws += align64f(HDIM * HDIM / 2);

    for (int l = 0; l < N_LAYERS; ++l) {
        const float* hin = (l == 0) ? x : hbuf;

        hipMemsetAsync(mkey, 0, sizeof(unsigned) * (size_t)N_REL * N, stream);
        hipMemsetAsync(ssum, 0, sizeof(float) * (size_t)N_REL * N, stream);
        hipMemsetAsync(outs, 0, sizeof(float) * (size_t)N_REL * N * HDIM, stream);
        hipMemsetAsync(colsum, 0, sizeof(float) * N_REL * HDIM, stream);

        pack_w_kernel<<<1, 256, 0, stream>>>(W_proj + l * HDIM * HDIM, wt);
        pack_w_kernel<<<1, 256, 0, stream>>>(Wk + l * HDIM * HDIM, wkt);

        proj_wmma<<<N / 32, 256, 0, stream>>>(hin, wt, b_proj + l * HDIM, hproj);

        alphas_kernel<<<(N + 7) / 8, 256, 0, stream>>>(
            hproj, a_src + l * N_REL * HDIM, a_dst + l * N_REL * HDIM, asrc, adst, N);

        dim3 ge((E + 255) / 256, N_REL);
        emax_kernel<<<ge, 256, 0, stream>>>(ei, asrc, adst, mkey, N, E);
        esum_kernel<<<ge, 256, 0, stream>>>(ei, asrc, adst, mkey, ssum, pbuf, N, E);

        dim3 ga((E + 7) / 8, N_REL);
        agg_kernel<<<ga, 256, 0, stream>>>(ei, hproj, pbuf, ssum, outs, N, E);

        long tot4 = (long)N_REL * N * HDIM / 4;
        relu4_kernel<<<(int)((tot4 + 255) / 256), 256, 0, stream>>>((float4*)outs, tot4);

        for (int r = 0; r < N_REL; ++r)
            semantic_wmma<<<N / 32, 256, 0, stream>>>(
                outs + (long)r * N * HDIM, wkt, bk + l * HDIM, colsum + r * HDIM);

        beta_kernel<<<1, 64, 0, stream>>>(colsum, q + l * HDIM, beta, N);

        long h4 = (long)N * HDIM / 4;
        combine_kernel<<<(int)((h4 + 255) / 256), 256, 0, stream>>>(
            (const float4*)outs, beta, (float4*)hbuf, h4, (l < N_LAYERS - 1) ? 1 : 0);
    }

    hipMemsetAsync(pooled, 0, sizeof(float) * N_GRAPHS * HDIM, stream);
    hipMemsetAsync(counts, 0, sizeof(float) * N_GRAPHS, stream);
    pool_kernel<<<(N + 7) / 8, 256, 0, stream>>>(hbuf, batch, pooled, counts, N);
    classify_kernel<<<1, 512, 0, stream>>>(pooled, counts, W_out, b_out, (float*)d_out);
}
